// GraphConvolution_90546500534486
// MI455X (gfx1250) — matrix operand order repaired and checked by reading, not compiled, not run on hardware
//
#include <hip/hip_runtime.h>

// MI455X / gfx1250, wave32. GEMM via v_wmma_f32_16x16x32_f16 (f32 accum),
// B matrix staged in LDS (bank-conflict-padded) and batch-loaded per K-tile
// so DS latency overlaps WMMA execution. h kept as f16 in workspace
// (25.6 MB -> L2 resident for the gather phase).

typedef __attribute__((ext_vector_type(16))) _Float16 v16h;
typedef __attribute__((ext_vector_type(8)))  _Float16 v8h;
typedef __attribute__((ext_vector_type(4)))  _Float16 v4h;
typedef __attribute__((ext_vector_type(8)))  float    v8f;
typedef __attribute__((ext_vector_type(4)))  float    v4f;

#define N_NODES 100000
#define KDEG    32
#define DIN     256
#define DOUT    128
#define ROW_TILES (N_NODES / 16)   // 6250 exactly
#define LDSP    (DIN + 8)          // LDS pitch in halves: 528B rows -> conflict-free B reads

// ---------------------------------------------------------------------------
// Micro-kernel: W[256,128] f32 row-major -> Wt[128,256] f16 (column-major W),
// so B fragments are contiguous per lane. 128 blocks x 256 threads.
// ---------------------------------------------------------------------------
__global__ __launch_bounds__(256)
void transpose_convert_W(const float* __restrict__ W, _Float16* __restrict__ Wt) {
    const int n = blockIdx.x;     // 0..127 (DOUT)
    const int k = threadIdx.x;    // 0..255 (DIN)
    Wt[(size_t)n * DIN + k] = (_Float16)W[(size_t)k * DOUT + n];
}

// ---------------------------------------------------------------------------
// GEMM + bias + ReLU: h16[N, 128] = relu(feats @ W + b), f16 output.
// 8 waves/block; each wave computes a 16-row band x full DOUT.
// B fragments come from LDS; A streams from HBM with in-register f32->f16.
// ---------------------------------------------------------------------------
__global__ __launch_bounds__(256)
void gemm_bias_relu(const float* __restrict__ feats,
                    const _Float16* __restrict__ Wt,
                    const float* __restrict__ bias,
                    _Float16* __restrict__ h16) {
    __shared__ _Float16 WtLds[DOUT * LDSP];        // ~67.6 KB of the 320 KB/WGP

    const int tid  = threadIdx.x;
    const int lane = tid & 31;
    const int wave = tid >> 5;

    // ---- cooperative fill: Wt (64 KB f16) -> padded LDS, all waves join ----
#pragma unroll
    for (int it = 0; it < 8; ++it) {
        const int c  = tid + it * 256;             // 0..2047 chunk id
        const int n  = c >> 4;                     // row (0..127)
        const int kc = (c & 15) * 16;              // 16-half chunk within row
        const v8h* src = (const v8h*)(Wt + (size_t)n * DIN + kc);
        v8h s0 = src[0];
        v8h s1 = src[1];
        *(v8h*)&WtLds[n * LDSP + kc]     = s0;     // 16B-aligned ds stores
        *(v8h*)&WtLds[n * LDSP + kc + 8] = s1;
    }
    __syncthreads();

    const int rowTile = blockIdx.x * 8 + wave;     // wave-uniform
    if (rowTile >= ROW_TILES) return;              // whole-wave exit after barrier

    const int row0 = rowTile * 16;
    const int lmod = lane & 15;                    // A: M index / B,C: N index
    const int lhi  = lane >> 4;                    // 0 or 1

    const float* aRow = feats + (size_t)(row0 + lmod) * DIN;

    v8f acc[8] = {};                               // 8 N-tiles x 8 VGPRs

    for (int k0 = 0; k0 < DIN; k0 += 32) {
        // ---- A fragment: 16-bit A 16x32 layout ----
        // lane<16: K = k0+0..7 (elems 0..7), k0+16..23 (elems 8..15)
        // lane>=16: +8 on both chunks. Contiguous 16B-aligned f32 reads.
        const float* ap = aRow + k0 + 8 * lhi;
        v4f a0 = *(const v4f*)(ap +  0);
        v4f a1 = *(const v4f*)(ap +  4);
        v4f a2 = *(const v4f*)(ap + 16);
        v4f a3 = *(const v4f*)(ap + 20);
        v16h aF;
#pragma unroll
        for (int i = 0; i < 4; ++i) {
            aF[i]      = (_Float16)a0[i];
            aF[4 + i]  = (_Float16)a1[i];
            aF[8 + i]  = (_Float16)a2[i];
            aF[12 + i] = (_Float16)a3[i];
        }

        // ---- batch-load all 8 B fragments for this K-tile (16 ds_load_b128),
        //      then run the 8 WMMAs: DS latency overlaps matrix execution.
        // 16-bit B 32x16 layout: lane = column n, lanes 0-15 hold K k0..k0+7, k0+16..k0+23,
        // lanes 16-31 hold K k0+8..k0+15, k0+24..k0+31 (same order as A).
        v16h bF[8];
#pragma unroll
        for (int t = 0; t < 8; ++t) {
            const _Float16* bp = &WtLds[(t * 16 + lmod) * LDSP + k0 + 8 * lhi];
            v8h b0 = *(const v8h*)bp;              // 16B ds_load_b128, conflict-free
            v8h b1 = *(const v8h*)(bp + 16);
            bF[t] = __builtin_shufflevector(b0, b1,
                        0, 1, 2, 3, 4, 5, 6, 7, 8, 9, 10, 11, 12, 13, 14, 15);
        }
#pragma unroll
        for (int t = 0; t < 8; ++t) {
            acc[t] = __builtin_amdgcn_wmma_f32_16x16x32_f16(
                         false, aF, false, bF[t], (short)0, acc[t], false, false);
        }
    }

    // ---- bias + relu + f32->f16 store ----
    // C layout: acc[t][r] holds (row = row0 + r + 8*lhi, col = t*16 + lmod)
#pragma unroll
    for (int t = 0; t < 8; ++t) {
        const float bv = bias[t * 16 + lmod];
#pragma unroll
        for (int r = 0; r < 8; ++r) {
            float v = acc[t][r] + bv;
            v = v > 0.0f ? v : 0.0f;
            const int row = row0 + r + 8 * lhi;
            h16[(size_t)row * DOUT + t * 16 + lmod] = (_Float16)v;
        }
    }
}

// ---------------------------------------------------------------------------
// Gather + mean: out[n, :] = mean over 32 neighbors of h16[nbr, :].
// One wave per node; lane owns 4 columns; 256B coalesced row reads from L2.
// ---------------------------------------------------------------------------
__global__ __launch_bounds__(256)
void gather_mean(const _Float16* __restrict__ h16,
                 const int* __restrict__ edges,
                 float* __restrict__ out) {
    const int lane = threadIdx.x & 31;
    const int wave = threadIdx.x >> 5;
    const int node = blockIdx.x * 8 + wave;
    if (node >= N_NODES) return;

    // lane L holds neighbor index #L for this node (one coalesced b32 load)
    const int myNbr = edges[(size_t)node * KDEG + lane];

    float ax = 0.0f, ay = 0.0f, az = 0.0f, aw = 0.0f;
#pragma unroll 8
    for (int j = 0; j < KDEG; ++j) {
        const int nbr = __shfl(myNbr, j, 32);       // wave-uniform broadcast
        const v4h hv = *(const v4h*)(h16 + (size_t)nbr * DOUT + lane * 4);  // 8B/lane
        ax += (float)hv[0];
        ay += (float)hv[1];
        az += (float)hv[2];
        aw += (float)hv[3];
    }
    const float s = 1.0f / (float)KDEG;
    float4 r;
    r.x = ax * s; r.y = ay * s; r.z = az * s; r.w = aw * s;
    *(float4*)(out + (size_t)node * DOUT + lane * 4) = r;   // 16B/lane coalesced
}

// ---------------------------------------------------------------------------
extern "C" void kernel_launch(void* const* d_in, const int* in_sizes, int n_in,
                              void* d_out, int out_size, void* d_ws, size_t ws_size,
                              hipStream_t stream) {
    const float* feats = (const float*)d_in[0];   // [N, 256] f32
    const int*   edges = (const int*)d_in[1];     // [N, 32]  i32
    const float* W     = (const float*)d_in[2];   // [256, 128] f32
    const float* bias  = (const float*)d_in[3];   // [128] f32
    float*       out   = (float*)d_out;           // [N, 128] f32

    // Workspace layout: h16 first (25.6 MB), then Wt (64 KB), both 32B aligned.
    _Float16* h16 = (_Float16*)d_ws;
    _Float16* Wt  = (_Float16*)((char*)d_ws + (size_t)N_NODES * DOUT * sizeof(_Float16));

    transpose_convert_W<<<DOUT, DIN, 0, stream>>>(W, Wt);
    gemm_bias_relu<<<(ROW_TILES + 7) / 8, 256, 0, stream>>>(feats, Wt, bias, h16);
    gather_mean<<<N_NODES / 8, 256, 0, stream>>>(h16, edges, out);
}
